// MTLModel_31593779429829
// MI455X (gfx1250) — matrix operand order repaired and checked by reading, not compiled, not run on hardware
//
#include <hip/hip_runtime.h>
#include <hip/hip_bf16.h>

// MI455X / gfx1250, wave32. WMMA bf16 16x16x32, f32 accumulate.

typedef __bf16 bf16_t;
typedef __attribute__((ext_vector_type(16))) __bf16 v16bf;
typedef __attribute__((ext_vector_type(8)))  float  v8f;

#define B_    1024
#define T_    512
#define L1    511     // conv1 valid output length
#define P1    256     // after maxpool(pad=1)
#define C1    128
#define C2    64
#define L2    128     // after maxpool(pad=0)  == LSTM sequence length
#define GATES 256
#define EPS_  1e-5f

__device__ __forceinline__ float sigmoidf_(float x) { return 1.0f / (1.0f + expf(-x)); }

// ---- WMMA fragment loaders (CDNA5 16-bit layouts, wave32) -------------------
// A-matrix 16x32 (MxK), row-major source with leading dim ld, tile at column kbase.
// lanes 0-15: M=lane, K = {0..7, 16..23}; lanes 16-31: M=lane-16, K = {8..15, 24..31}.
// Each half is 8 contiguous bf16 -> compiler merges into 128-bit loads.
__device__ __forceinline__ v16bf load_a_frag_bf16(const bf16_t* base, int ld, int kbase) {
  int lane  = threadIdx.x & 31;
  int m     = lane & 15;
  int khalf = (lane < 16) ? 0 : 8;
  const bf16_t* row = base + m * ld + kbase + khalf;
  v16bf a;
#pragma unroll
  for (int e = 0; e < 8; ++e) a[e] = row[e];          // K = khalf + e
#pragma unroll
  for (int e = 8; e < 16; ++e) a[e] = row[8 + e];     // K = khalf + 8 + e
  return a;
}

// B-matrix 32x16 (KxN) from a pre-transposed bf16 W^T stored row-major [K][N]:
// lane supplies N = nbase + lane%16; K = {0..7, 16..23} (lanes 0-15), {8..15, 24..31} (lanes 16-31).
__device__ __forceinline__ v16bf load_b_frag_bf16t(const bf16_t* Wt, int ldn, int nbase, int kbase) {
  int lane = threadIdx.x & 31;
  int h = lane >> 4;
  const bf16_t* col = Wt + (size_t)kbase * ldn + nbase + (lane & 15);
  v16bf b;
#pragma unroll
  for (int e = 0; e < 16; ++e) b[e] = col[(size_t)((e < 8) ? (8 * h + e) : (16 + 8 * h + (e - 8))) * ldn];
  return b;
}

// ---- K0: zero BN-stat accumulators -----------------------------------------
__global__ void k_zero_stats(float* __restrict__ stats) {
  int i = blockIdx.x * blockDim.x + threadIdx.x;
  if (i < 1024) stats[i] = 0.0f;
}

// ---- prep: one-time weight transpose + f32->bf16 convert --------------------
// w2 [64][128] -> w2t [128][64];  wih/whh [256][64] -> wiht/whht [64][256]
__global__ void k_prep_weights(const float* __restrict__ w2,
                               const float* __restrict__ wih, const float* __restrict__ whh,
                               bf16_t* __restrict__ w2t, bf16_t* __restrict__ wiht,
                               bf16_t* __restrict__ whht) {
  int i = blockIdx.x * blockDim.x + threadIdx.x;   // 0 .. 16383
  if (i < C1 * C2) {
    int k = i / C2, n = i % C2;
    w2t[i] = (bf16_t)w2[(size_t)n * C1 + k];
  }
  if (i < 64 * GATES) {
    int k = i / GATES, n = i % GATES;
    wiht[i] = (bf16_t)wih[(size_t)n * 64 + k];
    whht[i] = (bf16_t)whh[(size_t)n * 64 + k];
  }
}

// ---- K1: conv1 + per-channel sum/sumsq (pass 1 of training-mode BN) --------
__global__ void k_conv1_stats(const float* __restrict__ x, const float* __restrict__ w,
                              const float* __restrict__ bias, float* __restrict__ stats) {
  __shared__ float xls[T_ * 2];
  __shared__ float rs[256], rq[256];
  int b = blockIdx.x, tid = threadIdx.x;
  const float* xp = x + (size_t)b * T_ * 2;
  for (int i = tid; i < T_ * 2; i += 256) xls[i] = xp[i];
  __syncthreads();
  int c = tid & 127, half = tid >> 7;
  float w00 = w[c * 4 + 0], w01 = w[c * 4 + 1], w10 = w[c * 4 + 2], w11 = w[c * 4 + 3];
  float bc = bias[c];
  float s = 0.f, q = 0.f;
  for (int p = half; p < L1; p += 2) {
    float y = w00 * xls[p * 2] + w01 * xls[(p + 1) * 2] +
              w10 * xls[p * 2 + 1] + w11 * xls[(p + 1) * 2 + 1] + bc;
    s += y; q += y * y;
  }
  rs[tid] = s; rq[tid] = q;
  __syncthreads();
  if (half == 0) {
    atomicAdd(&stats[c],       rs[tid] + rs[tid + 128]);
    atomicAdd(&stats[128 + c], rq[tid] + rq[tid + 128]);
  }
}

// ---- finalize BN: scale = g*rsqrt(var+eps), shift = b - mean*scale ----------
__global__ void k_finalize_bn(const float* __restrict__ sum, const float* __restrict__ sumsq,
                              const float* __restrict__ g, const float* __restrict__ bb,
                              float* __restrict__ scale, float* __restrict__ shift,
                              int C, float inv_count) {
  int t = blockIdx.x * blockDim.x + threadIdx.x;
  if (t < C) {
    float mean = sum[t] * inv_count;
    float var  = sumsq[t] * inv_count - mean * mean;
    float sc   = g[t] * rsqrtf(var + EPS_);
    scale[t] = sc;
    shift[t] = bb[t] - mean * sc;
  }
}

// ---- K3: conv1 (recompute) + BN1 + ReLU + maxpool(pad=1) -> h1 bf16 [B*P1,C1]
__global__ void k_conv1_bn_pool(const float* __restrict__ x, const float* __restrict__ w,
                                const float* __restrict__ bias,
                                const float* __restrict__ scale, const float* __restrict__ shift,
                                bf16_t* __restrict__ h1) {
  __shared__ float xls[T_ * 2];
  int b = blockIdx.x, tid = threadIdx.x;
  const float* xp = x + (size_t)b * T_ * 2;
  for (int i = tid; i < T_ * 2; i += 256) xls[i] = xp[i];
  __syncthreads();
  int c = tid & 127, half = tid >> 7;
  float w00 = w[c * 4 + 0], w01 = w[c * 4 + 1], w10 = w[c * 4 + 2], w11 = w[c * 4 + 3];
  float bc = bias[c], sc = scale[c], sh = shift[c];
  auto bnrelu = [&](int p) {
    float y = w00 * xls[p * 2] + w01 * xls[(p + 1) * 2] +
              w10 * xls[p * 2 + 1] + w11 * xls[(p + 1) * 2 + 1] + bc;
    return fmaxf(y * sc + sh, 0.0f);
  };
  for (int po = half; po < P1; po += 2) {
    float v;
    if (po == 0) v = bnrelu(0);                                     // (-inf pad, y0)
    else         v = fmaxf(bnrelu(2 * po - 1), bnrelu(2 * po));
    h1[((size_t)b * P1 + po) * C1 + c] = (bf16_t)v;
  }
}

// ---- K4: conv2 (1x1) as WMMA GEMM [M=B*P1, N=64, K=128] + bias + BN2 stats --
// A tile (32x128 bf16) staged in LDS once per block; B from bf16 W^T.
__global__ void __launch_bounds__(256)
k_conv2_gemm(const bf16_t* __restrict__ h1, const bf16_t* __restrict__ w2t,
             const float* __restrict__ b2, float* __restrict__ y2,
             float* __restrict__ s2sum, float* __restrict__ s2sq) {
  __shared__ bf16_t As[32 * C1];                      // 8 KB
  int tid = threadIdx.x;
  int wave = tid >> 5, lane = tid & 31;
  int mw = wave >> 2, nw = wave & 3;                  // 2 M-subtiles x 4 N-subtiles
  int row0 = blockIdx.x * 32;
  {  // coalesced 128-bit staging: 8192 B = 512 uint4, 2 per thread
    const uint4* src = (const uint4*)(h1 + (size_t)row0 * C1);
    uint4* dst = (uint4*)As;
    dst[tid]       = src[tid];
    dst[tid + 256] = src[tid + 256];
  }
  __syncthreads();
  const bf16_t* abase = As + mw * 16 * C1;
  v8f acc = {};
#pragma unroll
  for (int kt = 0; kt < 4; ++kt) {
    v16bf a  = load_a_frag_bf16(abase, C1, kt * 32);
    v16bf bb = load_b_frag_bf16t(w2t, C2, nw * 16, kt * 32);
    acc = __builtin_amdgcn_wmma_f32_16x16x32_bf16(false, a, false, bb, (short)0, acc, false, false);
  }
  int n    = (lane & 15) + nw * 16;                   // C/D layout: channel fixed per lane
  int mofs = (lane >> 4) << 3;
  float bn = b2[n];
  float ps = 0.f, pq = 0.f;
#pragma unroll
  for (int r = 0; r < 8; ++r) {
    float v = acc[r] + bn;
    y2[((size_t)row0 + mw * 16 + r + mofs) * C2 + n] = v;
    ps += v; pq += v * v;
  }
  atomicAdd(&s2sum[n], ps);
  atomicAdd(&s2sq[n], pq);
}

// ---- K6: BN2 + ReLU + maxpool(pad=0) -> seq bf16 [B, L2, C2] ----------------
__global__ void k_bn2_pool(const float* __restrict__ y2, const float* __restrict__ scale,
                           const float* __restrict__ shift, bf16_t* __restrict__ seq) {
  int b = blockIdx.x, tid = threadIdx.x;
  for (int idx = tid; idx < L2 * C2; idx += 256) {
    int p = idx >> 6, c = idx & 63;
    float sc = scale[c], sh = shift[c];
    float a = fmaxf(y2[((size_t)b * P1 + 2 * p)     * C2 + c] * sc + sh, 0.0f);
    float d = fmaxf(y2[((size_t)b * P1 + 2 * p + 1) * C2 + c] * sc + sh, 0.0f);
    seq[((size_t)b * L2 + p) * C2 + c] = (bf16_t)fmaxf(a, d);
  }
}

// ---- K7: LSTM, 16 samples per block, WMMA for x@Wih^T and h@Whh^T -----------
__global__ void __launch_bounds__(256)
k_lstm(const bf16_t* __restrict__ seq,
       const bf16_t* __restrict__ wiht, const bf16_t* __restrict__ whht,
       const float* __restrict__ bih, const float* __restrict__ bhh,
       float* __restrict__ shared_out) {
  __shared__ bf16_t hs[16 * 64];
  __shared__ bf16_t xs[16 * 64];
  __shared__ float  gates[16][GATES];
  int tid  = threadIdx.x;
  int wave = tid >> 5, lane = tid & 31;
  int s0   = blockIdx.x * 16;
  int nbase = wave * 32;                              // this wave's 32 gate columns

  // Preload weight fragments: Wih^T and Whh^T (bf16, [64 x 256]). 8 frags/wave.
  v16bf fih[2][2], fhh[2][2];
#pragma unroll
  for (int kt = 0; kt < 2; ++kt)
#pragma unroll
    for (int nt = 0; nt < 2; ++nt) {
      fih[kt][nt] = load_b_frag_bf16t(wiht, GATES, nbase + nt * 16, kt * 32);
      fhh[kt][nt] = load_b_frag_bf16t(whht, GATES, nbase + nt * 16, kt * 32);
    }
  int ncol   = nbase + (lane & 15);
  float bias0 = bih[ncol] + bhh[ncol];
  float bias1 = bih[ncol + 16] + bhh[ncol + 16];
  int mofs = (lane >> 4) << 3;
  int nl   = lane & 15;

  for (int i = tid; i < 16 * 64; i += 256) hs[i] = (bf16_t)0.0f;
  int m  = tid >> 4;                                  // sample within tile (elementwise)
  int u0 = (tid & 15) * 4;                            // 4 hidden units per thread
  float creg[4] = {0.f, 0.f, 0.f, 0.f};
  float hsum[4] = {0.f, 0.f, 0.f, 0.f};
  __syncthreads();

  for (int t = 0; t < L2; ++t) {
    // stage x_t tile [16 x 64] into LDS
    {
      const bf16_t* src = seq + (((size_t)(s0 + m)) * L2 + t) * C2 + u0;
      bf16_t* dst = &xs[m * 64 + u0];
      dst[0] = src[0]; dst[1] = src[1]; dst[2] = src[2]; dst[3] = src[3];
    }
    __syncthreads();

    // gates[16,256] = x_t @ Wih^T + h @ Whh^T   (8 WMMAs per wave per step)
    v8f acc0 = {}, acc1 = {};
#pragma unroll
    for (int kt = 0; kt < 2; ++kt) {
      v16bf ax = load_a_frag_bf16(xs, 64, kt * 32);
      v16bf ah = load_a_frag_bf16(hs, 64, kt * 32);
      acc0 = __builtin_amdgcn_wmma_f32_16x16x32_bf16(false, ax, false, fih[kt][0], (short)0, acc0, false, false);
      acc0 = __builtin_amdgcn_wmma_f32_16x16x32_bf16(false, ah, false, fhh[kt][0], (short)0, acc0, false, false);
      acc1 = __builtin_amdgcn_wmma_f32_16x16x32_bf16(false, ax, false, fih[kt][1], (short)0, acc1, false, false);
      acc1 = __builtin_amdgcn_wmma_f32_16x16x32_bf16(false, ah, false, fhh[kt][1], (short)0, acc1, false, false);
    }
#pragma unroll
    for (int r = 0; r < 8; ++r) {
      gates[r + mofs][nbase + nl]      = acc0[r] + bias0;
      gates[r + mofs][nbase + 16 + nl] = acc1[r] + bias1;
    }
    __syncthreads();

    // elementwise LSTM cell (torch gate order i,f,g,o); c stays in registers
#pragma unroll
    for (int j = 0; j < 4; ++j) {
      int u = u0 + j;
      float gi = gates[m][u], gf = gates[m][64 + u];
      float gg = gates[m][128 + u], go = gates[m][192 + u];
      creg[j] = sigmoidf_(gf) * creg[j] + sigmoidf_(gi) * tanhf(gg);
      float h = sigmoidf_(go) * tanhf(creg[j]);
      hsum[j] += h;
      hs[m * 64 + u] = (bf16_t)h;
    }
    __syncthreads();
  }
#pragma unroll
  for (int j = 0; j < 4; ++j)
    shared_out[((size_t)(s0 + m)) * C2 + u0 + j] = hsum[j] * (1.0f / (float)L2);
}

// ---- K8: task-routed heads (tiny batched GEMV; VALU) ------------------------
__global__ void k_heads(const float* __restrict__ shared_in, const int* __restrict__ task_ids,
                        const float* __restrict__ W1, const float* __restrict__ b1,
                        const float* __restrict__ W2, const float* __restrict__ b2,
                        const float* __restrict__ Wo, const float* __restrict__ bo,
                        float* __restrict__ out) {
  __shared__ float sv[64], h1v[128], h2v[64], red[64];
  int b = blockIdx.x, tid = threadIdx.x;
  int task = task_ids[b];
  if (tid < 64) sv[tid] = shared_in[(size_t)b * 64 + tid];
  __syncthreads();
  {
    const float* wr = W1 + ((size_t)task * 128 + tid) * 64;
    float acc = b1[(size_t)task * 128 + tid];
#pragma unroll 8
    for (int k = 0; k < 64; ++k) acc += wr[k] * sv[k];
    h1v[tid] = fmaxf(acc, 0.0f);
  }
  __syncthreads();
  if (tid < 64) {
    const float* wr = W2 + ((size_t)task * 64 + tid) * 128;
    float acc = b2[(size_t)task * 64 + tid];
#pragma unroll 8
    for (int k = 0; k < 128; ++k) acc += wr[k] * h1v[k];
    h2v[tid] = fmaxf(acc, 0.0f);
  }
  __syncthreads();
  if (tid < 64) red[tid] = Wo[(size_t)task * 64 + tid] * h2v[tid];
  __syncthreads();
  for (int s = 32; s > 0; s >>= 1) {
    if (tid < s) red[tid] += red[tid + s];
    __syncthreads();
  }
  if (tid == 0) out[b] = red[0] + bo[task];
}

// -----------------------------------------------------------------------------
extern "C" void kernel_launch(void* const* d_in, const int* in_sizes, int n_in,
                              void* d_out, int out_size, void* d_ws, size_t ws_size,
                              hipStream_t stream) {
  (void)in_sizes; (void)n_in; (void)out_size; (void)ws_size;
  const float* x        = (const float*)d_in[0];
  const int*   task_ids = (const int*)d_in[1];
  const float* conv1_w  = (const float*)d_in[2];
  const float* conv1_b  = (const float*)d_in[3];
  const float* bn1_g    = (const float*)d_in[4];
  const float* bn1_b    = (const float*)d_in[5];
  const float* conv2_w  = (const float*)d_in[6];
  const float* conv2_b  = (const float*)d_in[7];
  const float* bn2_g    = (const float*)d_in[8];
  const float* bn2_b    = (const float*)d_in[9];
  const float* wih      = (const float*)d_in[10];
  const float* whh      = (const float*)d_in[11];
  const float* bih      = (const float*)d_in[12];
  const float* bhh      = (const float*)d_in[13];
  const float* W1       = (const float*)d_in[14];
  const float* b1       = (const float*)d_in[15];
  const float* W2       = (const float*)d_in[16];
  const float* b2       = (const float*)d_in[17];
  const float* Wo       = (const float*)d_in[18];
  const float* bo       = (const float*)d_in[19];
  float* out = (float*)d_out;

  // workspace layout
  char* ws = (char*)d_ws;
  float* stats = (float*)ws;  // [0:128) bn1_sum [128:256) bn1_sq [256:384) sc1 [384:512) sh1
                              // [512:576) bn2_sum [576:640) bn2_sq [640:704) sc2 [704:768) sh2
  size_t off = 4096;
  bf16_t* h1   = (bf16_t*)(ws + off);  off += (size_t)B_ * P1 * C1 * sizeof(bf16_t); // 64 MB
  float*  y2   = (float*)(ws + off);   off += (size_t)B_ * P1 * C2 * sizeof(float);  // 64 MB
  bf16_t* sq   = (bf16_t*)(ws + off);  off += (size_t)B_ * L2 * C2 * sizeof(bf16_t); // 16 MB
  float*  sh   = (float*)(ws + off);   off += (size_t)B_ * C2 * sizeof(float);       // 256 KB
  bf16_t* w2t  = (bf16_t*)(ws + off);  off += (size_t)C1 * C2 * sizeof(bf16_t);      // 16 KB
  bf16_t* wiht = (bf16_t*)(ws + off);  off += (size_t)64 * GATES * sizeof(bf16_t);   // 32 KB
  bf16_t* whht = (bf16_t*)(ws + off);                                                // 32 KB

  k_zero_stats<<<4, 256, 0, stream>>>(stats);
  k_prep_weights<<<64, 256, 0, stream>>>(conv2_w, wih, whh, w2t, wiht, whht);
  k_conv1_stats<<<B_, 256, 0, stream>>>(x, conv1_w, conv1_b, stats);
  k_finalize_bn<<<1, 128, 0, stream>>>(stats, stats + 128, bn1_g, bn1_b,
                                       stats + 256, stats + 384, 128,
                                       1.0f / (float)(B_ * L1));
  k_conv1_bn_pool<<<B_, 256, 0, stream>>>(x, conv1_w, conv1_b,
                                          stats + 256, stats + 384, h1);
  k_conv2_gemm<<<(B_ * P1) / 32, 256, 0, stream>>>(h1, w2t, conv2_b, y2,
                                                   stats + 512, stats + 576);
  k_finalize_bn<<<1, 64, 0, stream>>>(stats + 512, stats + 576, bn2_g, bn2_b,
                                      stats + 640, stats + 704, 64,
                                      1.0f / (float)(B_ * P1));
  k_bn2_pool<<<B_, 256, 0, stream>>>(y2, stats + 640, stats + 704, sq);
  k_lstm<<<B_ / 16, 256, 0, stream>>>(sq, wiht, whht, bih, bhh, sh);
  k_heads<<<B_, 128, 0, stream>>>(sh, task_ids, W1, b1, W2, b2, Wo, bo, out);
}
